// SubGraph_86749749444720
// MI455X (gfx1250) — matrix operand order repaired and checked by reading, not compiled, not run on hardware
//
#include <hip/hip_runtime.h>

typedef __attribute__((ext_vector_type(16))) _Float16 v16h;
typedef __attribute__((ext_vector_type(8)))  float    v8f;

#define NPTS 600000
#define KCL  30000

union V16H {
    v16h     v;
    int4     q[2];
    _Float16 h[16];
};

// Order-preserving monotone map f32 -> u32 so unsigned atomicMax == float max.
__device__ __forceinline__ unsigned encMax(float f) {
    unsigned u = __float_as_uint(f);
    return (u & 0x80000000u) ? ~u : (u | 0x80000000u);
}
__device__ __forceinline__ float decMax(unsigned u) {
    unsigned b = (u & 0x80000000u) ? (u & 0x7FFFFFFFu) : ~u;
    return __uint_as_float(b);
}
#define ENC_NEGINF 0x007FFFFFu   // encMax(-inf)

// ---------------------------------------------------------------------------
// Weight prep: f32 row-major [K][64] -> f16 transposed [n][Kpad] (B-fragment
// friendly: two 16B reads, 32B apart, per lane per v_wmma B operand).
// ---------------------------------------------------------------------------
__global__ void prep_kernel(
    const float* __restrict__ W10, const float* __restrict__ W11,
    const float* __restrict__ W12, const float* __restrict__ W20,
    const float* __restrict__ W21, const float* __restrict__ W22,
    const float* __restrict__ WL,
    _Float16* __restrict__ wt1_0, _Float16* __restrict__ wt1_1,
    _Float16* __restrict__ wt1_2, _Float16* __restrict__ wt2_0,
    _Float16* __restrict__ wt2_1, _Float16* __restrict__ wt2_2,
    _Float16* __restrict__ wtL)
{
    int tid = blockIdx.x * 256 + threadIdx.x;
    if (tid >= 64 * 128) return;
    int n = tid >> 7, k = tid & 127;
    if (k < 32) wt1_0[n * 32 + k] = (k < 10) ? (_Float16)W10[k * 64 + n] : (_Float16)0.f;
    wt1_1[n * 128 + k] = (_Float16)W11[k * 64 + n];
    wt1_2[n * 128 + k] = (_Float16)W12[k * 64 + n];
    if (k < 64) {
        wt2_0[n * 64 + k] = (_Float16)W20[k * 64 + n];
        wt2_1[n * 64 + k] = (_Float16)W21[k * 64 + n];
        wt2_2[n * 64 + k] = (_Float16)W22[k * 64 + n];
    }
    wtL[n * 128 + k] = (_Float16)WL[k * 64 + n];
}

__global__ void seg_init_kernel(unsigned* __restrict__ segA)
{
    int i = blockIdx.x * 256 + threadIdx.x;
    if (i < KCL * 64) segA[i] = ENC_NEGINF;
}

// Decode atomic table to f16 gather table and re-arm for the next layer.
__global__ void seg_finalize_kernel(unsigned* __restrict__ segA,
                                    _Float16* __restrict__ segF)
{
    int i = blockIdx.x * 256 + threadIdx.x;
    if (i < KCL * 64) {
        segF[i] = (_Float16)decMax(segA[i]);
        segA[i] = ENC_NEGINF;
    }
}

// ---------------------------------------------------------------------------
// Fused MLP layer: x@W1+b1 -> LN -> ReLU -> @W2+b2 -> LN -> ReLU
//   -> y (f16, global) and encoded atomic segment-max.
// One wave = one 16-row tile; 4 waves per block; N % 64 == 0 so no tails.
// LAYER 0 reads raw x [N,10] (K padded to 32); LAYER 1/2 read the virtual
// concat (y_prev f16 | segF[cluster] f16) as K-chunks, no [N,128] buffer.
// ---------------------------------------------------------------------------
template<int LAYER>
__global__ __launch_bounds__(128) void mlp_kernel(
    const float* __restrict__ x0,
    const _Float16* __restrict__ yin,
    const _Float16* __restrict__ segF,
    const int* __restrict__ cluster,
    const _Float16* __restrict__ wt1, const float* __restrict__ b1,
    const float* __restrict__ g1, const float* __restrict__ bt1,
    const _Float16* __restrict__ wt2, const float* __restrict__ b2,
    const float* __restrict__ g2, const float* __restrict__ bt2,
    _Float16* __restrict__ yout, unsigned* __restrict__ segA)
{
    constexpr int KP1 = (LAYER == 0) ? 32 : 128;
    const int wave = threadIdx.x >> 5;
    const int lane = threadIdx.x & 31;
    const int mrow = lane & 15;     // row within tile (A row index, C column index)
    const int hw   = lane >> 4;     // half-wave
    const int base = blockIdx.x * 64 + wave * 16;
    const int row  = base + mrow;

    __shared__ float ldsF[4][16 * 66];                      // per-wave f32 tile
    __shared__ __align__(16) _Float16 ldsH[4][16 * 64];     // per-wave f16 tile
    float*    lf = ldsF[wave];
    _Float16* lh = ldsH[wave];

    const int c = cluster[row];

    // ---------------- GEMM1: acc = x @ W1 + b1 ----------------
    v8f acc[4];
#pragma unroll
    for (int t = 0; t < 4; ++t) {
        float bb = b1[16 * t + mrow];
#pragma unroll
        for (int r = 0; r < 8; ++r) acc[t][r] = bb;
    }

#pragma unroll
    for (int kk = 0; kk < KP1 / 32; ++kk) {
        V16H a;
        if constexpr (LAYER == 0) {
            const int kb = hw * 8;
#pragma unroll
            for (int e = 0; e < 8; ++e) {
                int k0 = kb + e;
                int k1 = kb + 16 + e;
                a.h[e]     = (k0 < 10) ? (_Float16)x0[row * 10 + k0] : (_Float16)0.f;
                a.h[8 + e] = (k1 < 10) ? (_Float16)x0[row * 10 + k1] : (_Float16)0.f;
            }
        } else {
            const _Float16* src = (kk < 2) ? (yin + row * 64 + 32 * kk)
                                           : (segF + c * 64 + 32 * (kk - 2));
            a.q[0] = *(const int4*)(src + hw * 8);
            a.q[1] = *(const int4*)(src + hw * 8 + 16);
        }
#pragma unroll
        for (int t = 0; t < 4; ++t) {
            V16H b;
            const _Float16* wp = wt1 + (16 * t + mrow) * KP1 + 32 * kk + hw * 8;
            b.q[0] = ((const int4*)wp)[0];
            b.q[1] = ((const int4*)wp)[2];
            acc[t] = __builtin_amdgcn_wmma_f32_16x16x32_f16(
                false, a.v, false, b.v, (short)0, acc[t], false, false);
        }
    }

    // ---------------- LayerNorm1 + ReLU -> f16 LDS (A-layout friendly) ------
#pragma unroll
    for (int t = 0; t < 4; ++t)
#pragma unroll
        for (int r = 0; r < 8; ++r)
            lf[(hw * 8 + r) * 66 + 16 * t + mrow] = acc[t][r];
    __syncthreads();

    const int ch = hw * 32;
    float vv[32];
    float s = 0.f, sq = 0.f;
#pragma unroll
    for (int j = 0; j < 32; ++j) {
        float v = lf[mrow * 66 + ch + j];
        vv[j] = v; s += v; sq += v * v;
    }
    s  += __shfl_xor(s, 16);
    sq += __shfl_xor(sq, 16);
    float mean = s * 0.015625f;
    float inv  = rsqrtf(sq * 0.015625f - mean * mean + 1e-5f);
#pragma unroll
    for (int j = 0; j < 32; ++j) {
        float v = (vv[j] - mean) * inv * g1[ch + j] + bt1[ch + j];
        v = fmaxf(v, 0.f);
        lh[mrow * 64 + ch + j] = (_Float16)v;
    }
    __syncthreads();

    // ---------------- GEMM2: acc = h @ W2 + b2 ----------------
#pragma unroll
    for (int t = 0; t < 4; ++t) {
        float bb = b2[16 * t + mrow];
#pragma unroll
        for (int r = 0; r < 8; ++r) acc[t][r] = bb;
    }
#pragma unroll
    for (int kk = 0; kk < 2; ++kk) {
        V16H a;
        const _Float16* src = lh + mrow * 64 + 32 * kk + hw * 8;
        a.q[0] = *(const int4*)src;
        a.q[1] = *(const int4*)(src + 16);
#pragma unroll
        for (int t = 0; t < 4; ++t) {
            V16H b;
            const _Float16* wp = wt2 + (16 * t + mrow) * 64 + 32 * kk + hw * 8;
            b.q[0] = ((const int4*)wp)[0];
            b.q[1] = ((const int4*)wp)[2];
            acc[t] = __builtin_amdgcn_wmma_f32_16x16x32_f16(
                false, a.v, false, b.v, (short)0, acc[t], false, false);
        }
    }

    // ---------------- LayerNorm2 + ReLU -> y out + encoded segmax -----------
    __syncthreads();
#pragma unroll
    for (int t = 0; t < 4; ++t)
#pragma unroll
        for (int r = 0; r < 8; ++r)
            lf[(hw * 8 + r) * 66 + 16 * t + mrow] = acc[t][r];
    __syncthreads();

    s = 0.f; sq = 0.f;
#pragma unroll
    for (int j = 0; j < 32; ++j) {
        float v = lf[mrow * 66 + ch + j];
        vv[j] = v; s += v; sq += v * v;
    }
    s  += __shfl_xor(s, 16);
    sq += __shfl_xor(sq, 16);
    mean = s * 0.015625f;
    inv  = rsqrtf(sq * 0.015625f - mean * mean + 1e-5f);

    V16H o0, o1;
#pragma unroll
    for (int j = 0; j < 32; ++j) {
        float v = (vv[j] - mean) * inv * g2[ch + j] + bt2[ch + j];
        v = fmaxf(v, 0.f);
        if (j < 16) o0.h[j] = (_Float16)v; else o1.h[j - 16] = (_Float16)v;
        atomicMax(segA + c * 64 + ch + j, encMax(v));   // L2-resident table
    }
    int4* dst = (int4*)(yout + row * 64 + ch);
    dst[0] = o0.q[0]; dst[1] = o0.q[1];
    dst[2] = o1.q[0]; dst[3] = o1.q[1];
}

// ---------------------------------------------------------------------------
// Final: (y2 | segF[cluster]) @ lin_W + lin_b, encoded atomic segment-max.
// ---------------------------------------------------------------------------
__global__ __launch_bounds__(128) void final_kernel(
    const _Float16* __restrict__ yin, const _Float16* __restrict__ segF,
    const int* __restrict__ cluster,
    const _Float16* __restrict__ wtL, const float* __restrict__ lb,
    unsigned* __restrict__ segA)
{
    const int wave = threadIdx.x >> 5;
    const int lane = threadIdx.x & 31;
    const int mrow = lane & 15;
    const int hw   = lane >> 4;
    const int base = blockIdx.x * 64 + wave * 16;
    const int row  = base + mrow;
    const int c    = cluster[row];

    v8f acc[4];
#pragma unroll
    for (int t = 0; t < 4; ++t) {
        float bb = lb[16 * t + mrow];
#pragma unroll
        for (int r = 0; r < 8; ++r) acc[t][r] = bb;
    }
#pragma unroll
    for (int kk = 0; kk < 4; ++kk) {
        V16H a;
        const _Float16* src = (kk < 2) ? (yin + row * 64 + 32 * kk)
                                       : (segF + c * 64 + 32 * (kk - 2));
        a.q[0] = *(const int4*)(src + hw * 8);
        a.q[1] = *(const int4*)(src + hw * 8 + 16);
#pragma unroll
        for (int t = 0; t < 4; ++t) {
            V16H b;
            const _Float16* wp = wtL + (16 * t + mrow) * 128 + 32 * kk + hw * 8;
            b.q[0] = ((const int4*)wp)[0];
            b.q[1] = ((const int4*)wp)[2];
            acc[t] = __builtin_amdgcn_wmma_f32_16x16x32_f16(
                false, a.v, false, b.v, (short)0, acc[t], false, false);
        }
    }

    int cl[8];
#pragma unroll
    for (int r = 0; r < 8; ++r) cl[r] = cluster[base + hw * 8 + r];
#pragma unroll
    for (int t = 0; t < 4; ++t)
#pragma unroll
        for (int r = 0; r < 8; ++r)
            atomicMax(segA + cl[r] * 64 + 16 * t + mrow, encMax(acc[t][r]));
}

// One wave per cluster row: decode, L2-normalize, write d_out.
__global__ __launch_bounds__(32) void normalize_kernel(
    const unsigned* __restrict__ segA, float* __restrict__ out)
{
    const int rowk = blockIdx.x;
    const int lane = threadIdx.x;
    float v0 = decMax(segA[rowk * 64 + lane]);
    float v1 = decMax(segA[rowk * 64 + 32 + lane]);
    float s = v0 * v0 + v1 * v1;
#pragma unroll
    for (int m = 16; m >= 1; m >>= 1) s += __shfl_xor(s, m);
    float sc = 1.f / fmaxf(sqrtf(s), 1e-12f);
    out[rowk * 64 + lane]      = v0 * sc;
    out[rowk * 64 + 32 + lane] = v1 * sc;
}

// ---------------------------------------------------------------------------
extern "C" void kernel_launch(void* const* d_in, const int* in_sizes, int n_in,
                              void* d_out, int out_size, void* d_ws, size_t ws_size,
                              hipStream_t stream) {
    (void)in_sizes; (void)n_in; (void)out_size; (void)ws_size;
    // setup_inputs flatten order:
    // 0:x 1:cluster 2:n_clusters 3..10:layer0(W1,b1,g1,bt1,W2,b2,g2,bt2)
    // 11..18:layer1 19..26:layer2 27:lin_W 28:lin_b
    const float* x       = (const float*)d_in[0];
    const int*   cluster = (const int*)d_in[1];
    const float* W1[3] = {(const float*)d_in[3],  (const float*)d_in[11], (const float*)d_in[19]};
    const float* b1[3] = {(const float*)d_in[4],  (const float*)d_in[12], (const float*)d_in[20]};
    const float* g1[3] = {(const float*)d_in[5],  (const float*)d_in[13], (const float*)d_in[21]};
    const float* t1[3] = {(const float*)d_in[6],  (const float*)d_in[14], (const float*)d_in[22]};
    const float* W2[3] = {(const float*)d_in[7],  (const float*)d_in[15], (const float*)d_in[23]};
    const float* b2[3] = {(const float*)d_in[8],  (const float*)d_in[16], (const float*)d_in[24]};
    const float* g2[3] = {(const float*)d_in[9],  (const float*)d_in[17], (const float*)d_in[25]};
    const float* t2[3] = {(const float*)d_in[10], (const float*)d_in[18], (const float*)d_in[26]};
    const float* linW = (const float*)d_in[27];
    const float* linb = (const float*)d_in[28];

    // Workspace carve (all offsets 16B aligned).
    char* w = (char*)d_ws;
    unsigned*  segA = (unsigned*)w;  w += (size_t)KCL * 64 * 4;   // atomic table
    _Float16*  segF = (_Float16*)w;  w += (size_t)KCL * 64 * 2;   // f16 gather table
    _Float16*  yA   = (_Float16*)w;  w += (size_t)NPTS * 64 * 2;
    _Float16*  yB   = (_Float16*)w;  w += (size_t)NPTS * 64 * 2;
    _Float16*  wt1_0 = (_Float16*)w; w += 64 * 32 * 2;
    _Float16*  wt1_1 = (_Float16*)w; w += 64 * 128 * 2;
    _Float16*  wt1_2 = (_Float16*)w; w += 64 * 128 * 2;
    _Float16*  wt2_0 = (_Float16*)w; w += 64 * 64 * 2;
    _Float16*  wt2_1 = (_Float16*)w; w += 64 * 64 * 2;
    _Float16*  wt2_2 = (_Float16*)w; w += 64 * 64 * 2;
    _Float16*  wtL   = (_Float16*)w; w += 64 * 128 * 2;

    const int segGrid = (KCL * 64) / 256;   // 7500
    const int mlpGrid = NPTS / 64;          // 9375

    prep_kernel<<<32, 256, 0, stream>>>(W1[0], W1[1], W1[2], W2[0], W2[1], W2[2],
                                        linW, wt1_0, wt1_1, wt1_2,
                                        wt2_0, wt2_1, wt2_2, wtL);
    seg_init_kernel<<<segGrid, 256, 0, stream>>>(segA);

    mlp_kernel<0><<<mlpGrid, 128, 0, stream>>>(x, nullptr, nullptr, cluster,
        wt1_0, b1[0], g1[0], t1[0], wt2_0, b2[0], g2[0], t2[0], yA, segA);
    seg_finalize_kernel<<<segGrid, 256, 0, stream>>>(segA, segF);

    mlp_kernel<1><<<mlpGrid, 128, 0, stream>>>(nullptr, yA, segF, cluster,
        wt1_1, b1[1], g1[1], t1[1], wt2_1, b2[1], g2[1], t2[1], yB, segA);
    seg_finalize_kernel<<<segGrid, 256, 0, stream>>>(segA, segF);

    mlp_kernel<2><<<mlpGrid, 128, 0, stream>>>(nullptr, yB, segF, cluster,
        wt1_2, b1[2], g1[2], t1[2], wt2_2, b2[2], g2[2], t2[2], yA, segA);
    seg_finalize_kernel<<<segGrid, 256, 0, stream>>>(segA, segF);

    final_kernel<<<mlpGrid, 128, 0, stream>>>(yA, segF, cluster, wtL, linb, segA);
    normalize_kernel<<<KCL, 32, 0, stream>>>(segA, (float*)d_out);
}
